// CompilerFriendlyAttention_38259568673104
// MI455X (gfx1250) — matrix operand order repaired and checked by reading, not compiled, not run on hardware
//
#include <hip/hip_runtime.h>

#define D_MODEL 1024
#define N_HEADS 16
#define D_HEAD  64
#define SEQ_T   2048
#define BATCH   4
#define NTOK    (BATCH * SEQ_T)          // 8192
// softmax in base-2 domain: t = s * SCALE * log2(e); softmax_2(t) == softmax_e(s*SCALE)
#define SL2E    (0.125f * 1.44269504088896340736f)

#define AS1 __attribute__((address_space(1)))
#define AS3 __attribute__((address_space(3)))

#if defined(__has_builtin)
#if __has_builtin(__builtin_amdgcn_global_load_async_to_lds_b128)
#define ASYNC_LDS_OK 1
#endif
#endif

typedef __attribute__((ext_vector_type(16))) __bf16 v16bf;
typedef __attribute__((ext_vector_type(8)))  float  v8f;
typedef int v4i __attribute__((vector_size(16)));

union Frag16 {
    v16bf v;
    uint4 q[2];
};
static_assert(sizeof(Frag16) == 32, "frag size");

__device__ __forceinline__ unsigned short f2bf(float f) {
    unsigned int u = __builtin_bit_cast(unsigned int, f);
    unsigned int r = u + 0x7FFFu + ((u >> 16) & 1u);   // round-to-nearest-even
    return (unsigned short)(r >> 16);
}

// A-fragment (16x32 bf16, MxK): lanes 0-15 hold rows 0-15 with K=0..15;
// lanes 16-31 hold rows 0-15 with K=16..31. src row-major, ld in elems.
__device__ __forceinline__ v16bf load_frag_a(const unsigned short* base,
                                             int ld, int row0, int k0) {
    const int lane = threadIdx.x & 31;
    const int hi   = lane >> 4;
    const int r    = lane & 15;
    const unsigned short* p = base + (size_t)(row0 + r) * ld + k0 + hi * 16;
    Frag16 f;
    f.q[0] = *(const uint4*)(p);        // K = k0 + hi*16 + 0..7
    f.q[1] = *(const uint4*)(p + 8);    // K = k0 + hi*16 + 8..15
    return f.v;
}

// B-fragment (32x16 bf16, KxN): column n = row (n0+lane%16) of a row-major
// [N,K] source; lanes 0-15 carry K=k0..k0+15, lanes 16-31 carry K=k0+16..k0+31.
__device__ __forceinline__ v16bf load_frag_b(const unsigned short* __restrict__ base,
                                             int ld, int n0, int k0) {
    const int lane = threadIdx.x & 31;
    const int hi   = lane >> 4;
    const int c    = lane & 15;
    const unsigned short* p = base + (size_t)(n0 + c) * ld + k0 + hi * 16;
    Frag16 f;
    f.q[0] = *(const uint4*)(p);
    f.q[1] = *(const uint4*)(p + 8);
    return f.v;
}

__device__ __forceinline__ v8f wmma_bf16(v16bf a, v16bf b, v8f c) {
    return __builtin_amdgcn_wmma_f32_16x16x32_bf16(
        /*neg_a=*/false, a, /*neg_b=*/false, b,
        /*c_mod=*/(short)0, c, /*reuse_a=*/false, /*reuse_b=*/false);
}

// Stage a 32x32 bf16 A tile (2 KiB) into LDS cooperatively (128 threads,
// 16 bytes each) using the gfx1250 async global->LDS DMA path when available.
__device__ __forceinline__ void stage_a_tile(const unsigned short* __restrict__ A,
                                             unsigned short* lds_dst,
                                             int K, int m0, int k0) {
    const int tid = threadIdx.x;            // 0..127
    const int row = tid >> 2;               // 0..31
    const int chk = (tid & 3) * 8;          // 0,8,16,24 (elements)
    const unsigned short* gp = A + (size_t)(m0 + row) * K + k0 + chk;
    unsigned short* lp = lds_dst + row * 32 + chk;
#ifdef ASYNC_LDS_OK
    AS1 const void* g1 = (AS1 const void*)gp;   // addrspacecast to global
    AS3 void*       l3 = (AS3 void*)lp;         // addrspacecast to LDS
    __builtin_amdgcn_global_load_async_to_lds_b128(
        (AS1 v4i*)g1, (AS3 v4i*)l3, 0, 0);
#else
    *(uint4*)lp = *(const uint4*)gp;
#endif
}

__device__ __forceinline__ void wait_stage() {
#ifdef ASYNC_LDS_OK
#if __has_builtin(__builtin_amdgcn_s_wait_asynccnt)
    __builtin_amdgcn_s_wait_asynccnt(0);
#else
    asm volatile("s_wait_asynccnt 0" ::: "memory");
#endif
#endif
    __syncthreads();
}

// ---------------------------------------------------------------- conversion
__global__ void cvt_f32_bf16(const float* __restrict__ in,
                             unsigned short* __restrict__ out, int n) {
    int i = blockIdx.x * blockDim.x + threadIdx.x;
    if (i < n) out[i] = f2bf(in[i]);
}

// -------------------------------------------------- GEMM: C = A @ W^T (bf16)
// A: [M,K] bf16 row-major. W: [Nout,K] bf16 row-major (torch Linear layout).
// Block = 128 threads (4 waves). All waves share one 32-row A tile staged into
// LDS (double-buffered async DMA); each wave owns a 64-col slice -> 32x64 tile,
// 8 WMMAs per 32-wide k-step with 2x B-fragment reuse.
// MODE 0: bf16 C row-major.  MODE 1: V transposed per head.  MODE 2: f32+bias.
template <int MODE>
__global__ void __launch_bounds__(128)
gemm32(const unsigned short* __restrict__ A,
       const unsigned short* __restrict__ W,
       const float* __restrict__ bias,
       void* __restrict__ Cout, int M, int K, int Nout) {
    __shared__ unsigned short atile[2][32 * 32];
    const int lane = threadIdx.x & 31;
    const int wave = threadIdx.x >> 5;
    const int m0 = blockIdx.x * 32;
    const int n0 = (blockIdx.y * 4 + wave) * 64;
    const int hi = lane >> 4, cl = lane & 15;

    v8f accA[4] = {{}, {}, {}, {}};
    v8f accB[4] = {{}, {}, {}, {}};

    stage_a_tile(A, &atile[0][0], K, m0, 0);
    int buf = 0;
    for (int k0 = 0; k0 < K; k0 += 32) {
        wait_stage();                               // atile[buf] is ready
        if (k0 + 32 < K)
            stage_a_tile(A, &atile[buf ^ 1][0], K, m0, k0 + 32);
        // B fragments for this k-step (streamed from global/L2)
        v16bf b0 = load_frag_b(W, K, n0,      k0);
        v16bf b1 = load_frag_b(W, K, n0 + 16, k0);
        v16bf b2 = load_frag_b(W, K, n0 + 32, k0);
        v16bf b3 = load_frag_b(W, K, n0 + 48, k0);
        if (k0 + 32 < K) {                          // prefetch next B lines
            __builtin_prefetch(W + (size_t)(n0 + cl) * K + k0 + 32, 0, 1);
            __builtin_prefetch(W + (size_t)(n0 + 16 + cl) * K + k0 + 32, 0, 1);
            __builtin_prefetch(W + (size_t)(n0 + 32 + cl) * K + k0 + 32, 0, 1);
            __builtin_prefetch(W + (size_t)(n0 + 48 + cl) * K + k0 + 32, 0, 1);
        }
        // A fragments from LDS
        v16bf a0 = load_frag_a(&atile[buf][0], 32, 0, 0);
        v16bf a1 = load_frag_a(&atile[buf][0], 32, 16, 0);
        accA[0] = wmma_bf16(a0, b0, accA[0]);
        accB[0] = wmma_bf16(a1, b0, accB[0]);
        accA[1] = wmma_bf16(a0, b1, accA[1]);
        accB[1] = wmma_bf16(a1, b1, accB[1]);
        accA[2] = wmma_bf16(a0, b2, accA[2]);
        accB[2] = wmma_bf16(a1, b2, accB[2]);
        accA[3] = wmma_bf16(a0, b3, accA[3]);
        accB[3] = wmma_bf16(a1, b3, accB[3]);
        buf ^= 1;
    }

#pragma unroll
    for (int j = 0; j < 4; ++j) {
        const int o = n0 + 16 * j + cl;
        if (MODE == 0) {
            unsigned short* C = (unsigned short*)Cout;
#pragma unroll
            for (int r = 0; r < 8; ++r) {
                C[(size_t)(m0 + hi * 8 + r) * Nout + o]      = f2bf(accA[j][r]);
                C[(size_t)(m0 + 16 + hi * 8 + r) * Nout + o] = f2bf(accB[j][r]);
            }
        } else if (MODE == 1) {
            unsigned short* Vt = (unsigned short*)Cout;
            const int h = o >> 6, dd = o & 63;
#pragma unroll
            for (int r = 0; r < 8; ++r) {
                int row = m0 + hi * 8 + r;
                int bb = row >> 11, t = row & (SEQ_T - 1);
                Vt[(size_t)((bb * N_HEADS + h) * D_HEAD + dd) * SEQ_T + t] = f2bf(accA[j][r]);
                row += 16;
                bb = row >> 11; t = row & (SEQ_T - 1);
                Vt[(size_t)((bb * N_HEADS + h) * D_HEAD + dd) * SEQ_T + t] = f2bf(accB[j][r]);
            }
        } else {
            float* Out = (float*)Cout;
            const float bv = bias[o];
#pragma unroll
            for (int r = 0; r < 8; ++r) {
                Out[(size_t)(m0 + hi * 8 + r) * Nout + o]      = accA[j][r] + bv;
                Out[(size_t)(m0 + 16 + hi * 8 + r) * Nout + o] = accB[j][r] + bv;
            }
        }
    }
}

// ------------------------------------------------------- flash attention
// Q,K in [NTOK, D] bf16 (head h occupies cols h*64..h*64+63).
// Vt in [(b*H+h)*64 + dd][T] bf16. Ao out in [NTOK, D] bf16.
// One wave per 16-query tile; online softmax over 32-key blocks.
__global__ void __launch_bounds__(128)
attn_kernel(const unsigned short* __restrict__ Q,
            const unsigned short* __restrict__ Km,
            const unsigned short* __restrict__ Vt,
            unsigned short* __restrict__ Ao) {
    __shared__ unsigned short pbuf[4][16 * 32];
    const int lane = threadIdx.x & 31;
    const int wave = threadIdx.x >> 5;
    const int wid  = blockIdx.x * 4 + wave;
    const int qt = wid & 127;
    const int h  = (wid >> 7) & (N_HEADS - 1);
    const int b  = wid >> 11;
    const int hi = lane >> 4, cl = lane & 15;

    const int m0 = b * SEQ_T + qt * 16;     // first query token row
    const int ch = h * D_HEAD;              // column offset of this head
    const int vrow0 = (b * N_HEADS + h) * D_HEAD;
    unsigned short* pb = &pbuf[wave][0];

    const v16bf qa0 = load_frag_a(Q, D_MODEL, m0, ch);
    const v16bf qa1 = load_frag_a(Q, D_MODEL, m0, ch + 32);

    v8f acc0 = {}, acc1 = {}, acc2 = {}, acc3 = {};
    float mrow[8], lrow[8];
#pragma unroll
    for (int r = 0; r < 8; ++r) { mrow[r] = -3.0e38f; lrow[r] = 0.0f; }

    for (int t0 = 0; t0 < SEQ_T; t0 += 32) {
        // ---- scores S[16 x 32] = Q_tile @ K_block^T  (4 WMMAs)
        v8f s0 = {}, s1 = {};
        {
            v16bf k00 = load_frag_b(Km, D_MODEL, b * SEQ_T + t0,      ch);
            v16bf k01 = load_frag_b(Km, D_MODEL, b * SEQ_T + t0,      ch + 32);
            v16bf k10 = load_frag_b(Km, D_MODEL, b * SEQ_T + t0 + 16, ch);
            v16bf k11 = load_frag_b(Km, D_MODEL, b * SEQ_T + t0 + 16, ch + 32);
            s0 = wmma_bf16(qa0, k00, s0);
            s0 = wmma_bf16(qa1, k01, s0);
            s1 = wmma_bf16(qa0, k10, s1);
            s1 = wmma_bf16(qa1, k11, s1);
        }
        // ---- V fragments issued early: global latency hides under softmax
        v16bf v0 = load_frag_b(Vt, SEQ_T, vrow0 +  0, t0);
        v16bf v1 = load_frag_b(Vt, SEQ_T, vrow0 + 16, t0);
        v16bf v2 = load_frag_b(Vt, SEQ_T, vrow0 + 32, t0);
        v16bf v3 = load_frag_b(Vt, SEQ_T, vrow0 + 48, t0);
        // ---- online softmax (base-2 domain), per accumulator row r
#pragma unroll
        for (int r = 0; r < 8; ++r) {
            float a  = s0[r] * SL2E;
            float c  = s1[r] * SL2E;
            float bm = fmaxf(a, c);
#pragma unroll
            for (int off = 1; off < 16; off <<= 1)
                bm = fmaxf(bm, __shfl_xor(bm, off, 32));   // stays in 16-lane half
            float mn   = fmaxf(mrow[r], bm);
            float corr = exp2f(mrow[r] - mn);
            float p0   = exp2f(a - mn);
            float p1   = exp2f(c - mn);
            float rs   = p0 + p1;
#pragma unroll
            for (int off = 1; off < 16; off <<= 1)
                rs += __shfl_xor(rs, off, 32);
            lrow[r] = lrow[r] * corr + rs;
            mrow[r] = mn;
            acc0[r] *= corr; acc1[r] *= corr; acc2[r] *= corr; acc3[r] *= corr;
            // write P (C-layout -> row-major 16x32 bf16 in LDS)
            pb[(hi * 8 + r) * 32 + cl]      = f2bf(p0);
            pb[(hi * 8 + r) * 32 + cl + 16] = f2bf(p1);
        }
        asm volatile("s_wait_dscnt 0" ::: "memory");
        // ---- O += P @ V_block  (4 WMMAs over d = 64)
        v16bf pa = load_frag_a(pb, 32, 0, 0);
        acc0 = wmma_bf16(pa, v0, acc0);
        acc1 = wmma_bf16(pa, v1, acc1);
        acc2 = wmma_bf16(pa, v2, acc2);
        acc3 = wmma_bf16(pa, v3, acc3);
    }
    // ---- normalize and store in [token, D] layout for the output projection
#pragma unroll
    for (int r = 0; r < 8; ++r) {
        float inv = 1.0f / lrow[r];
        size_t base = (size_t)(m0 + hi * 8 + r) * D_MODEL + ch + cl;
        Ao[base +  0] = f2bf(acc0[r] * inv);
        Ao[base + 16] = f2bf(acc1[r] * inv);
        Ao[base + 32] = f2bf(acc2[r] * inv);
        Ao[base + 48] = f2bf(acc3[r] * inv);
    }
}

// --------------------------------------------------------------- launcher
extern "C" void kernel_launch(void* const* d_in, const int* in_sizes, int n_in,
                              void* d_out, int out_size, void* d_ws, size_t ws_size,
                              hipStream_t stream) {
    const float* x  = (const float*)d_in[0];
    const float* Wq = (const float*)d_in[1];
    const float* Wk = (const float*)d_in[2];
    const float* Wv = (const float*)d_in[3];
    const float* Wo = (const float*)d_in[4];
    const float* bo = (const float*)d_in[5];

    char* ws = (char*)d_ws;
    const size_t XB  = (size_t)NTOK * D_MODEL * 2;     // 16 MiB
    const size_t WB  = (size_t)D_MODEL * D_MODEL * 2;  //  2 MiB
    unsigned short* xb = (unsigned short*)(ws);
    unsigned short* wq = (unsigned short*)(ws + XB);
    unsigned short* wk = (unsigned short*)(ws + XB + WB);
    unsigned short* wv = (unsigned short*)(ws + XB + 2 * WB);
    unsigned short* wo = (unsigned short*)(ws + XB + 3 * WB);
    unsigned short* qb = (unsigned short*)(ws + XB + 4 * WB);
    unsigned short* kb = (unsigned short*)(ws + 2 * XB + 4 * WB);
    unsigned short* vt = (unsigned short*)(ws + 3 * XB + 4 * WB);
    unsigned short* ao = (unsigned short*)(ws + 4 * XB + 4 * WB);

    const int nx = NTOK * D_MODEL;          // 8388608
    const int nw = D_MODEL * D_MODEL;       // 1048576
    cvt_f32_bf16<<<nx / 256, 256, 0, stream>>>(x,  xb, nx);
    cvt_f32_bf16<<<nw / 256, 256, 0, stream>>>(Wq, wq, nw);
    cvt_f32_bf16<<<nw / 256, 256, 0, stream>>>(Wk, wk, nw);
    cvt_f32_bf16<<<nw / 256, 256, 0, stream>>>(Wv, wv, nw);
    cvt_f32_bf16<<<nw / 256, 256, 0, stream>>>(Wo, wo, nw);

    dim3 ggrid(NTOK / 32, D_MODEL / 256);   // (256, 4)
    gemm32<0><<<ggrid, 128, 0, stream>>>(xb, wq, nullptr, qb, NTOK, D_MODEL, D_MODEL);
    gemm32<0><<<ggrid, 128, 0, stream>>>(xb, wk, nullptr, kb, NTOK, D_MODEL, D_MODEL);
    gemm32<1><<<ggrid, 128, 0, stream>>>(xb, wv, nullptr, vt, NTOK, D_MODEL, D_MODEL);

    const int nwaves = BATCH * N_HEADS * (SEQ_T / 16);  // 8192
    attn_kernel<<<nwaves / 4, 128, 0, stream>>>(qb, kb, vt, ao);

    gemm32<2><<<ggrid, 128, 0, stream>>>(ao, wo, bo, (float*)d_out,
                                         NTOK, D_MODEL, D_MODEL);
}
